// BoltzmannMachine_10007273799685
// MI455X (gfx1250) — matrix operand order repaired and checked by reading, not compiled, not run on hardware
//
#include <hip/hip_runtime.h>
#include <hip/hip_bf16.h>

// Boltzmann machine sequential Gibbs sweep, blocked with delta correction.
// N = 8192 units, 32 sequential blocks of 256 units each.
//  - bm_gemv_wmma: base[i] = w[perm[i]] . state via split-f16 WMMA
//                  (w = hi + lo, two chained v_wmma_f32_16x16x32_f16 per
//                  K-chunk -> fp32-accurate sums), 64-way split-K partials.
//  - bm_sweep:     sequential 256-step Gibbs scan inside one WGP using an LDS
//                  copy of the 256x256 gathered coupling sub-matrix.

#define NN 8192
#define BLK 256
#define NBLOCKS (NN / BLK)        // 32
#define KSLICES 64
#define KS_LEN (NN / KSLICES)     // 128 columns -> 4 K=32 WMMA steps
#define NTILES (BLK / 16)         // 16 row tiles
#define WAVES_PER_WG 8
#define WSUB_STRIDE 257           // 256 + 1 pad -> conflict-free column reads

typedef __attribute__((ext_vector_type(16))) _Float16 v16h;
typedef __attribute__((ext_vector_type(8)))  float    v8f;
typedef __attribute__((ext_vector_type(4)))  float    v4f;

// ---------------------------------------------------------------------------
// GEMV via WMMA: one wave per (row-tile, K-slice); 8 independent waves per
// workgroup, 1024 waves per block for memory-level parallelism.
// A fragment (16x32 f16): lane group g holds K = 8g..8g+7 and 16+8g..16+8g+7
// of row M = lane&15.  B fragment: state chunk broadcast to all 16 N columns
// (lane group g holds the same K values as A), so D[:,0] is the GEMV result.
// Split-f16: c += hi(w)*s; c += lo(w)*s  -> fp32-level accuracy.
// ---------------------------------------------------------------------------
__global__ __launch_bounds__(256) void bm_gemv_wmma(
    const float* __restrict__ w, const float* __restrict__ state,
    const int* __restrict__ perm, float* __restrict__ partial, int blk)
{
    const int lane = threadIdx.x & 31;
    const int wid  = threadIdx.x >> 5;
    const int waveid = blockIdx.x * WAVES_PER_WG + wid;   // 0..1023
    const int tile = waveid & (NTILES - 1);               // 0..15
    const int ks   = waveid >> 4;                         // 0..63
    const int g    = lane >> 4;                           // half-wave group
    const int M    = lane & 15;                           // row within tile

    const int row = perm[blk * BLK + tile * 16 + M];
    const float* __restrict__ wrow = w + (size_t)row * NN;
    const int kbase = ks * KS_LEN;

    // speculative prefetch of this row slice (global_prefetch_b8)
    __builtin_prefetch(wrow + kbase, 0, 1);

    v8f c = {0.f, 0.f, 0.f, 0.f, 0.f, 0.f, 0.f, 0.f};

#pragma unroll
    for (int kk = 0; kk < KS_LEN; kk += 32) {
        const int k0 = kbase + kk;

        // A: two contiguous 8-float chunks of this row
        v4f a0 = *(const v4f*)(wrow + k0 + 8 * g);
        v4f a1 = *(const v4f*)(wrow + k0 + 8 * g + 4);
        v4f a2 = *(const v4f*)(wrow + k0 + 16 + 8 * g);
        v4f a3 = *(const v4f*)(wrow + k0 + 16 + 8 * g + 4);
        // B: state values K = 8g..8g+7 and 16+8g..16+8g+7, same as A (broadcast)
        v4f b0 = *(const v4f*)(state + k0 + 8 * g);
        v4f b1 = *(const v4f*)(state + k0 + 8 * g + 4);
        v4f b2 = *(const v4f*)(state + k0 + 16 + 8 * g);
        v4f b3 = *(const v4f*)(state + k0 + 16 + 8 * g + 4);

        float af[16];
#pragma unroll
        for (int i = 0; i < 4; ++i) {
            af[i]      = a0[i];
            af[4 + i]  = a1[i];
            af[8 + i]  = a2[i];
            af[12 + i] = a3[i];
        }

        v16h ahi, alo, b;
#pragma unroll
        for (int i = 0; i < 16; ++i) {
            const _Float16 h = (_Float16)af[i];
            ahi[i] = h;
            alo[i] = (_Float16)(af[i] - (float)h);   // exact residual -> f16
        }
#pragma unroll
        for (int i = 0; i < 4; ++i) {
            b[i]      = (_Float16)b0[i];   // state is 0/1 -> exact in f16
            b[4 + i]  = (_Float16)b1[i];
            b[8 + i]  = (_Float16)b2[i];
            b[12 + i] = (_Float16)b3[i];
        }

        // D = hi*B + C ; D = lo*B + D   (v_wmma_f32_16x16x32_f16 x2)
        c = __builtin_amdgcn_wmma_f32_16x16x32_f16(
                false, ahi, false, b, (short)0, c, false, false);
        c = __builtin_amdgcn_wmma_f32_16x16x32_f16(
                false, alo, false, b, (short)0, c, false, false);
    }

    // Extract D column N==0: lane 0 holds rows 0..7 (vgpr r -> M=r),
    // lane 16 holds rows 8..15.
    const int pbase = (tile * KSLICES + ks) * 16;
    if (lane == 0) {
#pragma unroll
        for (int r = 0; r < 8; ++r) partial[pbase + r] = c[r];
    } else if (lane == 16) {
#pragma unroll
        for (int r = 0; r < 8; ++r) partial[pbase + 8 + r] = c[r];
    }
}

// ---------------------------------------------------------------------------
// Sequential block sweep. 256 threads: phase 1 gathers the 256x256 coupling
// sub-matrix into LDS and reduces split-K partials (deterministic fixed
// order); phase 2 runs the 256 dependent Gibbs steps on wave 0 only (no
// barriers needed inside the chain — single-wave LDS ordering).
// ---------------------------------------------------------------------------
__global__ __launch_bounds__(256) void bm_sweep(
    const float* __restrict__ w, const int* __restrict__ perm,
    const float* __restrict__ rand_u, const float* __restrict__ clampv,
    const float* __restrict__ Tptr, const float* __restrict__ partial,
    float* __restrict__ state, int blk)
{
    extern __shared__ float smem[];
    float* wsub = smem;                          // BLK * WSUB_STRIDE floats
    float* acc  = smem + BLK * WSUB_STRIDE;      // BLK
    float* ru   = acc + BLK;                     // BLK
    float* cl   = ru + BLK;                      // BLK
    float* st   = cl + BLK;                      // BLK
    int*   pblk = (int*)(st + BLK);              // BLK

    const int t    = threadIdx.x;
    const int gidx = blk * BLK + t;
    const int rowt = perm[gidx];
    pblk[t] = rowt;
    ru[t]   = rand_u[gidx];
    cl[t]   = clampv[rowt];
    st[t]   = state[rowt];

    // Reduce split-K partials in a fixed order (deterministic).
    float b = 0.f;
#pragma unroll 8
    for (int ks = 0; ks < KSLICES; ++ks)
        b += partial[((t >> 4) * KSLICES + ks) * 16 + (t & 15)];
    acc[t] = b;
    __syncthreads();

    // Gather w[perm_row][perm_col] sub-matrix into padded LDS tile.
    const float* __restrict__ wrow = w + (size_t)rowt * NN;
#pragma unroll 8
    for (int k = 0; k < BLK; ++k)
        wsub[t * WSUB_STRIDE + k] = wrow[pblk[k]];
    __syncthreads();

    if (t < 32) {
        const float invT = 1.0f / Tptr[0];
        for (int j = 0; j < BLK; ++j) {
            const float cj   = acc[j];                       // LDS broadcast
            const float p    = 1.0f / (1.0f + __expf(-cj * invT));
            const float newv = (ru[j] <= p) ? 1.0f : 0.0f;
            const float oldv = st[j];
            const float upd  = (cl[j] == 0.0f) ? newv : oldv;
            const float delta = upd - oldv;                  // uniform per wave
            if (t == 0) state[pblk[j]] = upd;
            if (delta != 0.0f) {
                // Rank-1 delta update of remaining accumulators.
#pragma unroll
                for (int m = 0; m < 8; ++m) {
                    const int tt = t + 32 * m;
                    if (tt > j)
                        acc[tt] += wsub[tt * WSUB_STRIDE + j] * delta;
                }
            }
        }
    }
}

extern "C" void kernel_launch(void* const* d_in, const int* in_sizes, int n_in,
                              void* d_out, int out_size, void* d_ws, size_t ws_size,
                              hipStream_t stream) {
    const float* w          = (const float*)d_in[0];
    const float* init_state = (const float*)d_in[1];
    const float* clampv     = (const float*)d_in[2];
    const float* Tptr       = (const float*)d_in[3];
    const int*   perm       = (const int*)d_in[4];
    const float* rand_u     = (const float*)d_in[5];

    float* state   = (float*)d_out;   // live state, becomes the final output
    float* partial = (float*)d_ws;    // NTILES * KSLICES * 16 = 16384 floats

    hipMemcpyAsync(state, init_state, (size_t)NN * sizeof(float),
                   hipMemcpyDeviceToDevice, stream);

    const size_t smem_bytes = (size_t)(BLK * WSUB_STRIDE + 5 * BLK) * sizeof(float);
    const int gemv_wgs = (NTILES * KSLICES) / WAVES_PER_WG;   // 128 workgroups

    for (int blk = 0; blk < NBLOCKS; ++blk) {
        bm_gemv_wmma<<<dim3(gemv_wgs), dim3(32 * WAVES_PER_WG), 0, stream>>>(
            w, state, perm, partial, blk);
        bm_sweep<<<dim3(1), dim3(BLK), smem_bytes, stream>>>(
            w, perm, rand_u, clampv, Tptr, partial, state, blk);
    }
}
